// HamHeadMeg_64793876628068
// MI455X (gfx1250) — matrix operand order repaired and checked by reading, not compiled, not run on hardware
//
#include <hip/hip_runtime.h>
#include <hip/hip_bf16.h>

typedef __attribute__((ext_vector_type(16))) _Float16 v16h;
typedef __attribute__((ext_vector_type(8)))  float    v8f;

#define LOG2F_C 0.6931471805599453f

__device__ __forceinline__ float sspf(float v) {
    // softplus(v) - log(2), numerically stable
    return fmaxf(v, 0.0f) + log1pf(expf(-fabsf(v))) - LOG2F_C;
}
__device__ __forceinline__ float sigmf(float v) { return 1.0f / (1.0f + expf(-v)); }

// order-preserving float<->uint encoding for atomicMax on signed floats
__device__ __forceinline__ unsigned enc_ord(float f) {
    unsigned u = __float_as_uint(f);
    return (u & 0x80000000u) ? ~u : (u | 0x80000000u);
}
__device__ __forceinline__ float dec_ord(unsigned u) {
    u = (u & 0x80000000u) ? (u & 0x7FFFFFFFu) : ~u;
    return __uint_as_float(u);
}

// ---------------------------------------------------------------- zero
__global__ void hhm_zero(float* p, long n) {
    long i = (long)blockIdx.x * blockDim.x + threadIdx.x;
    if (i < n) p[i] = 0.0f;
}

// ---------------------------------------------------------------- init: q vectors + segment accumulators
__global__ void hhm_init(const float* bih_v, const float* bhh_v,
                         const float* bih_e, const float* bhh_e,
                         float* qv, float* qe,
                         unsigned* maxV, unsigned* maxE,
                         float* sumV, float* sumE, float* rV, float* rE, int G) {
    int t = blockIdx.x * blockDim.x + threadIdx.x;
    if (t == 0) {
        // gates = bih + bhh ; split i,f,g,o (d=3); h=c=0 initially
        for (int k = 0; k < 3; ++k) {
            float gi = bih_v[k]     + bhh_v[k];
            float gg = bih_v[6 + k] + bhh_v[6 + k];
            float go = bih_v[9 + k] + bhh_v[9 + k];
            float c  = sigmf(gi) * tanhf(gg);
            qv[k] = sigmf(go) * tanhf(c);
            gi = bih_e[k]     + bhh_e[k];
            gg = bih_e[6 + k] + bhh_e[6 + k];
            go = bih_e[9 + k] + bhh_e[9 + k];
            c  = sigmf(gi) * tanhf(gg);
            qe[k] = sigmf(go) * tanhf(c);
        }
    }
    if (t < G) {
        maxV[t] = 0u;  // below enc_ord of any finite float
        maxE[t] = 0u;
        sumV[t] = 0.0f;
        sumE[t] = 0.0f;
        for (int j = 0; j < 3; ++j) { rV[t * 3 + j] = 0.0f; rE[t * 3 + j] = 0.0f; }
    }
}

// ---------------------------------------------------------------- edge kernel (WMMA f16 16x16x32, K=64)
// ze[e] = [x[src](16) | x[dst](16) | edge_attr(16) | state[bb](16)]  -> We1(64x6) -> ssp -> We2(6x3) -> ssp
__global__ void __launch_bounds__(256)
hhm_edge(const float* __restrict__ x, const int* __restrict__ edge_index,
         const float* __restrict__ edge_attr, const float* __restrict__ state,
         const int* __restrict__ bond_batch,
         const float* __restrict__ We1, const float* __restrict__ be1,
         const float* __restrict__ We2, const float* __restrict__ be2,
         const float* __restrict__ qe,
         float4* __restrict__ en4, float* __restrict__ acc_v,
         unsigned* __restrict__ maxE, long N, long E) {
    __shared__ _Float16 Wh[64 * 16];        // We1 padded to 16 cols, f16
    __shared__ float    We2s[18], be1s[6], be2s[3], qes[3];
    __shared__ _Float16 ze[8][16 * 64];     // per-wave A tile: 16 edges x K=64
    __shared__ float    ht[8][16 * 6];      // per-wave hidden activations

    const int t = threadIdx.x;
    // stage weights (f32 -> f16 for the WMMA operand)
    for (int idx = t; idx < 64 * 16; idx += 256) {
        int k = idx >> 4, n = idx & 15;
        Wh[idx] = (n < 6) ? (_Float16)We1[k * 6 + n] : (_Float16)0.0f;
    }
    if (t < 18) We2s[t] = We2[t];
    if (t < 6)  be1s[t] = be1[t];
    if (t < 3)  { be2s[t] = be2[t]; qes[t] = qe[t]; }
    __syncthreads();

    const int lane = t & 31;
    const int wave = t >> 5;
    const int col  = lane & 15;
    const int half = lane >> 4;
    const long tile = (long)blockIdx.x * 8 + wave;
    const long base = tile * 16;
    if (base >= E) return;                 // wave-uniform

    _Float16* zw = &ze[wave][0];
    float*    hw = &ht[wave][0];

    // ---- gather/stage ze tile: each lane loads features (lane) and (lane+32) per edge row
    for (int m = 0; m < 16; ++m) {
        long e = base + m; if (e >= E) e = E - 1;
        int src = edge_index[e];
        int dst = edge_index[E + e];
        int g   = bond_batch[e];
        float v1 = (lane < 16) ? x[(long)src * 16 + lane] : x[(long)dst * 16 + (lane - 16)];
        float v2 = (lane < 16) ? edge_attr[e * 16 + lane] : state[(long)g * 16 + (lane - 16)];
        zw[m * 64 + lane]      = (_Float16)v1;
        zw[m * 64 + 32 + lane] = (_Float16)v2;
    }
    __builtin_amdgcn_wave_barrier();
    asm volatile("s_wait_dscnt 0x0" ::: "memory");

    // ---- B fragments: 32x16, lanes 0-15 hold K=0..15 (elem k -> K=16*half+k)
    v16h b0, b1;
    #pragma unroll
    for (int k = 0; k < 16; ++k) {
        b0[k] = Wh[(16 * half + k) * 16 + col];
        b1[k] = Wh[(32 + 16 * half + k) * 16 + col];
    }
    // ---- A fragments: 16x32, row = lane&15; elem k -> K=16*half+k (matches B)
    const int row = lane & 15;
    v16h a0, a1;
    #pragma unroll
    for (int k = 0; k < 8; ++k) {
        a0[k]     = zw[row * 64 + 16 * half + k];
        a0[k + 8] = zw[row * 64 + 16 * half + 8 + k];
        a1[k]     = zw[row * 64 + 32 + 16 * half + k];
        a1[k + 8] = zw[row * 64 + 32 + 16 * half + 8 + k];
    }

    v8f c = {};
    c = __builtin_amdgcn_wmma_f32_16x16x32_f16(false, a0, false, b0, (short)0, c, false, false);
    c = __builtin_amdgcn_wmma_f32_16x16x32_f16(false, a1, false, b1, (short)0, c, false, false);

    // ---- activation + scatter hidden to LDS:  C elem r -> (M = r + 8*half, N = col)
    if (col < 6) {
        #pragma unroll
        for (int r = 0; r < 8; ++r) {
            int m = r + 8 * half;
            hw[m * 6 + col] = sspf(c[r] + be1s[col]);
        }
    }
    __builtin_amdgcn_wave_barrier();
    asm volatile("s_wait_dscnt 0x0" ::: "memory");

    // ---- layer 2 (6->3) + score + outputs; lane m handles edge base+m
    if (lane < 16) {
        long e = base + lane;
        if (e < E) {
            float h[6];
            #pragma unroll
            for (int k = 0; k < 6; ++k) h[k] = hw[lane * 6 + k];
            float o[3];
            #pragma unroll
            for (int j = 0; j < 3; ++j) {
                float s = be2s[j];
                #pragma unroll
                for (int k = 0; k < 6; ++k) s += h[k] * We2s[k * 3 + j];
                o[j] = sspf(s);
            }
            float sc = o[0] * qes[0] + o[1] * qes[1] + o[2] * qes[2];
            en4[e] = make_float4(o[0], o[1], o[2], sc);
            int dst = edge_index[E + e];
            atomicAdd(&acc_v[(long)dst * 4 + 0], o[0]);
            atomicAdd(&acc_v[(long)dst * 4 + 1], o[1]);
            atomicAdd(&acc_v[(long)dst * 4 + 2], o[2]);
            atomicAdd(&acc_v[(long)dst * 4 + 3], 1.0f);
            int g = bond_batch[e];
            atomicMax(&maxE[g], enc_ord(sc));
        }
    }
}

// ---------------------------------------------------------------- node kernel (35->6->3, VALU; trivial FLOPs)
__global__ void __launch_bounds__(256)
hhm_node(const float* __restrict__ x, const float* __restrict__ acc_v,
         const float* __restrict__ state, const int* __restrict__ batch,
         const float* __restrict__ Wv1, const float* __restrict__ bv1,
         const float* __restrict__ Wv2, const float* __restrict__ bv2,
         const float* __restrict__ qvw,
         float4* __restrict__ vn4, unsigned* __restrict__ maxV, long N) {
    __shared__ float W1s[35 * 6], b1s[6], W2s[18], b2s[3], qs[3];
    int t = threadIdx.x;
    for (int i = t; i < 210; i += 256) W1s[i] = Wv1[i];
    if (t < 6)  b1s[t] = bv1[t];
    if (t < 18) W2s[t] = Wv2[t];
    if (t < 3)  { b2s[t] = bv2[t]; qs[t] = qvw[t]; }
    __syncthreads();

    long n = (long)blockIdx.x * blockDim.x + t;
    if (n >= N) return;
    float z[35];
    #pragma unroll
    for (int k = 0; k < 16; ++k) z[k] = x[n * 16 + k];
    float4 a = ((const float4*)acc_v)[n];
    float cnt = fmaxf(a.w, 1.0f);
    z[16] = a.x / cnt; z[17] = a.y / cnt; z[18] = a.z / cnt;
    int g = batch[n];
    #pragma unroll
    for (int k = 0; k < 16; ++k) z[19 + k] = state[(long)g * 16 + k];

    float h[6];
    #pragma unroll
    for (int j = 0; j < 6; ++j) {
        float s = b1s[j];
        for (int k = 0; k < 35; ++k) s += z[k] * W1s[k * 6 + j];
        h[j] = sspf(s);
    }
    float o[3];
    #pragma unroll
    for (int j = 0; j < 3; ++j) {
        float s = b2s[j];
        #pragma unroll
        for (int k = 0; k < 6; ++k) s += h[k] * W2s[k * 3 + j];
        o[j] = sspf(s);
    }
    float sc = o[0] * qs[0] + o[1] * qs[1] + o[2] * qs[2];
    vn4[n] = make_float4(o[0], o[1], o[2], sc);
    atomicMax(&maxV[g], enc_ord(sc));
}

// ---------------------------------------------------------------- segment softmax pass 2
// feat.w = score; accumulates sum(exp) and sum(exp * feat.xyz) per segment.
// Segments are sorted -> wave-uniform fast path: shuffle-reduce, one atomic set per wave.
__global__ void __launch_bounds__(256)
hhm_pass2(const float4* __restrict__ feat, const int* __restrict__ seg,
          const unsigned* __restrict__ mx, float* __restrict__ sum,
          float* __restrict__ r, long M) {
    long i  = (long)blockIdx.x * blockDim.x + threadIdx.x;
    long ii = (i < M) ? i : (M - 1);
    float w = (i < M) ? 1.0f : 0.0f;
    float4 f = feat[ii];
    int g = seg[ii];
    float m  = dec_ord(mx[g]);
    float ex = expf(f.w - m) * w;
    float vx = ex * f.x, vy = ex * f.y, vz = ex * f.z;

    int g0 = __shfl(g, 0);
    if (__all(g == g0)) {
        #pragma unroll
        for (int s = 16; s >= 1; s >>= 1) {
            ex += __shfl_xor(ex, s);
            vx += __shfl_xor(vx, s);
            vy += __shfl_xor(vy, s);
            vz += __shfl_xor(vz, s);
        }
        if ((threadIdx.x & 31) == 0) {
            atomicAdd(&sum[g0], ex);
            atomicAdd(&r[g0 * 3 + 0], vx);
            atomicAdd(&r[g0 * 3 + 1], vy);
            atomicAdd(&r[g0 * 3 + 2], vz);
        }
    } else if (w > 0.0f) {
        atomicAdd(&sum[g], ex);
        atomicAdd(&r[g * 3 + 0], vx);
        atomicAdd(&r[g * 3 + 1], vy);
        atomicAdd(&r[g * 3 + 2], vz);
    }
}

// ---------------------------------------------------------------- final filters (G rows, tiny)
__device__ __forceinline__ float hhm_filt3(const float* v,
    const float* W1, const float* b1, const float* W2, const float* b2,
    const float* W3, const float* b3) {
    float h[6];
    #pragma unroll
    for (int j = 0; j < 6; ++j) {
        float s = b1[j];
        #pragma unroll
        for (int k = 0; k < 3; ++k) s += v[k] * W1[k * 6 + j];
        h[j] = sspf(s);
    }
    float h2[3];
    #pragma unroll
    for (int j = 0; j < 3; ++j) {
        float s = b2[j];
        #pragma unroll
        for (int k = 0; k < 6; ++k) s += h[k] * W2[k * 3 + j];
        h2[j] = sspf(s);
    }
    float o = b3[0];
    #pragma unroll
    for (int k = 0; k < 3; ++k) o += h2[k] * W3[k];
    return o;
}

__global__ void hhm_final(const float* sumV, const float* rV,
                          const float* sumE, const float* rE,
                          const float* Wf1v, const float* bf1v, const float* Wf2v, const float* bf2v,
                          const float* Wf3v, const float* bf3v,
                          const float* Wf1e, const float* bf1e, const float* Wf2e, const float* bf2e,
                          const float* Wf3e, const float* bf3e,
                          float* out, int G) {
    int g = blockIdx.x * blockDim.x + threadIdx.x;
    if (g >= G) return;
    float rv[3], re[3];
    float sv = sumV[g], se = sumE[g];
    for (int j = 0; j < 3; ++j) { rv[j] = rV[g * 3 + j] / sv; re[j] = rE[g * 3 + j] / se; }
    out[g]     = hhm_filt3(rv, Wf1v, bf1v, Wf2v, bf2v, Wf3v, bf3v);
    out[G + g] = hhm_filt3(re, Wf1e, bf1e, Wf2e, bf2e, Wf3e, bf3e);
}

// ================================================================ launch
extern "C" void kernel_launch(void* const* d_in, const int* in_sizes, int n_in,
                              void* d_out, int out_size, void* d_ws, size_t ws_size,
                              hipStream_t stream) {
    const float* x          = (const float*)d_in[0];
    const int*   edge_index = (const int*)  d_in[1];
    const float* edge_attr  = (const float*)d_in[2];
    const float* state      = (const float*)d_in[3];
    const int*   batch      = (const int*)  d_in[4];
    const int*   bond_batch = (const int*)  d_in[5];
    const float* We1  = (const float*)d_in[6];  const float* be1  = (const float*)d_in[7];
    const float* We2  = (const float*)d_in[8];  const float* be2  = (const float*)d_in[9];
    const float* Wv1  = (const float*)d_in[10]; const float* bv1  = (const float*)d_in[11];
    const float* Wv2  = (const float*)d_in[12]; const float* bv2  = (const float*)d_in[13];
    // d_in[14..17] = u-MLP: dead code in the reference output, skipped
    const float* bih_v = (const float*)d_in[20]; const float* bhh_v = (const float*)d_in[21];
    const float* bih_e = (const float*)d_in[24]; const float* bhh_e = (const float*)d_in[25];
    const float* Wf1v = (const float*)d_in[26]; const float* bf1v = (const float*)d_in[27];
    const float* Wf2v = (const float*)d_in[28]; const float* bf2v = (const float*)d_in[29];
    const float* Wf3v = (const float*)d_in[30]; const float* bf3v = (const float*)d_in[31];
    const float* Wf1e = (const float*)d_in[32]; const float* bf1e = (const float*)d_in[33];
    const float* Wf2e = (const float*)d_in[34]; const float* bf2e = (const float*)d_in[35];
    const float* Wf3e = (const float*)d_in[36]; const float* bf3e = (const float*)d_in[37];

    const long N = in_sizes[0] / 16;
    const long E = in_sizes[1] / 2;
    const int  G = in_sizes[3] / 16;

    // workspace layout (floats)
    float* ws = (float*)d_ws;
    float*    qv   = ws;
    float*    qe   = ws + 4;
    unsigned* maxV = (unsigned*)(ws + 8);
    unsigned* maxE = maxV + G;
    float*    sumV = (float*)(maxE + G);
    float*    sumE = sumV + G;
    float*    rV   = sumE + G;
    float*    rE   = rV + 3 * (size_t)G;
    size_t small_end = 8 + 10 * (size_t)G;
    size_t big = ((small_end + 1023) / 1024) * 1024;       // 4KB-align big arrays
    float*  acc_v = ws + big;                               // [N,4] sums + count
    float4* vn4   = (float4*)(acc_v + 4 * (size_t)N);       // [N] v_new + score
    float4* en4   = vn4 + N;                                // [E] e_new + score

    // 1) zero edge->node accumulators; init q consts + per-graph accumulators
    {
        long nz = 4 * N;
        hhm_zero<<<(unsigned)((nz + 255) / 256), 256, 0, stream>>>(acc_v, nz);
        hhm_init<<<(unsigned)((G + 255) / 256), 256, 0, stream>>>(
            bih_v, bhh_v, bih_e, bhh_e, qv, qe, maxV, maxE, sumV, sumE, rV, rE, G);
    }
    // 2) edge MLP via WMMA + scatter
    {
        long tiles  = (E + 15) / 16;
        long blocks = (tiles + 7) / 8;
        hhm_edge<<<(unsigned)blocks, 256, 0, stream>>>(
            x, edge_index, edge_attr, state, bond_batch,
            We1, be1, We2, be2, qe, en4, acc_v, maxE, N, E);
    }
    // 3) node MLP (needs acc_v complete)
    hhm_node<<<(unsigned)((N + 255) / 256), 256, 0, stream>>>(
        x, acc_v, state, batch, Wv1, bv1, Wv2, bv2, qv, vn4, maxV, N);
    // 4) segment softmax accumulation (needs segment maxima)
    hhm_pass2<<<(unsigned)((E + 255) / 256), 256, 0, stream>>>(en4, bond_batch, maxE, sumE, rE, E);
    hhm_pass2<<<(unsigned)((N + 255) / 256), 256, 0, stream>>>(vn4, batch, maxV, sumV, rV, N);
    // 5) filters -> h_ii, h_ij
    hhm_final<<<(unsigned)((G + 255) / 256), 256, 0, stream>>>(
        sumV, rV, sumE, rE,
        Wf1v, bf1v, Wf2v, bf2v, Wf3v, bf3v,
        Wf1e, bf1e, Wf2e, bf2e, Wf3e, bf3e,
        (float*)d_out, G);
    // 6) third output: edge_index passthrough (raw 32-bit copy)
    hipMemcpyAsync((char*)d_out + sizeof(float) * 2 * (size_t)G, d_in[1],
                   sizeof(int) * 2 * (size_t)E, hipMemcpyDeviceToDevice, stream);
}
